// MoEPositionwiseFFN_21114059227289
// MI455X (gfx1250) — matrix operand order repaired and checked by reading, not compiled, not run on hardware
//
#include <hip/hip_runtime.h>
#include <hip/hip_bf16.h>
#include <stdint.h>

// ---------------- problem constants (from reference) ----------------
#define D_MODEL 1024
#define H_FFN   4096
#define NEXP    8
#define TOPK    2

// ---------------- GEMM tiling ----------------
#define BM 128
#define BN 128
#define BK 32
#define LDT 40   // padded LDS row stride in bf16 elements (32 + 8 pad -> conflict-free 16B frag reads)

typedef __attribute__((ext_vector_type(16))) __bf16 bf16x16;
typedef __attribute__((ext_vector_type(2)))  __bf16 bf16x2;
typedef __attribute__((ext_vector_type(2)))  float  f32x2;
typedef __attribute__((ext_vector_type(8)))  float  f32x8;

union AFrag { bf16x16 v; uint4 q[2]; };

// HW bf16 conversion (RNE).
__device__ __forceinline__ unsigned short f2bf(float f) {
  union { __bf16 h; unsigned short u; } v;
  v.h = (__bf16)f;
  return v.u;
}
// Pack exactly (lo, hi) with one v_cvt_pk_bf16_f32 (vector fptrunc pattern).
__device__ __forceinline__ unsigned pack2bf(float lo, float hi) {
  f32x2 f; f.x = lo; f.y = hi;
  union { bf16x2 h; unsigned u; } v;
  v.h = __builtin_convertvector(f, bf16x2);
  return v.u;
}

// =====================================================================
// 1) Router: logits = x @ Wr + br, softmax, top-2 (ties -> lower index).
//    One wave32 per token.
// =====================================================================
__global__ __launch_bounds__(256)
void router_topk(const float* __restrict__ x, const float* __restrict__ Wr,
                 const float* __restrict__ br,
                 int* __restrict__ ids, float* __restrict__ gates, int N)
{
  int tok  = blockIdx.x * 8 + (threadIdx.x >> 5);
  int lane = threadIdx.x & 31;
  if (tok >= N) return;

  float acc[NEXP];
  #pragma unroll
  for (int e = 0; e < NEXP; ++e) acc[e] = 0.f;

  const float* xr = x + (size_t)tok * D_MODEL;
  for (int i = 0; i < D_MODEL / 32; ++i) {
    int d = i * 32 + lane;
    float xv = xr[d];
    const float4* w4 = (const float4*)&Wr[d * NEXP];
    float4 w0 = w4[0], w1 = w4[1];
    acc[0] += xv * w0.x; acc[1] += xv * w0.y; acc[2] += xv * w0.z; acc[3] += xv * w0.w;
    acc[4] += xv * w1.x; acc[5] += xv * w1.y; acc[6] += xv * w1.z; acc[7] += xv * w1.w;
  }
  #pragma unroll
  for (int e = 0; e < NEXP; ++e)
    #pragma unroll
    for (int off = 16; off > 0; off >>= 1)
      acc[e] += __shfl_xor(acc[e], off, 32);

  if (lane == 0) {
    float p[NEXP];
    float mx = -3.4e38f;
    #pragma unroll
    for (int e = 0; e < NEXP; ++e) { p[e] = acc[e] + br[e]; mx = fmaxf(mx, p[e]); }
    float s = 0.f;
    #pragma unroll
    for (int e = 0; e < NEXP; ++e) { p[e] = __expf(p[e] - mx); s += p[e]; }
    float inv = 1.f / s;
    #pragma unroll
    for (int e = 0; e < NEXP; ++e) p[e] *= inv;
    int i1 = 0;
    #pragma unroll
    for (int e = 1; e < NEXP; ++e) if (p[e] > p[i1]) i1 = e;
    int i2 = (i1 == 0) ? 1 : 0;
    #pragma unroll
    for (int e = 0; e < NEXP; ++e) if (e != i1 && p[e] > p[i2]) i2 = e;
    ids[tok * 2 + 0]   = i1;  gates[tok * 2 + 0] = p[i1];
    ids[tok * 2 + 1]   = i2;  gates[tok * 2 + 1] = p[i2];
  }
}

// =====================================================================
// 2) Zero dispatch slots (padded slots gather row 0; never combined).
// =====================================================================
__global__ void zero_disp(int* __restrict__ disp_tok, int n) {
  int i = blockIdx.x * blockDim.x + threadIdx.x;
  if (i < n) disp_tok[i] = 0;
}

// =====================================================================
// 3) Exact capacity ranking (reproduces stable lexsort((-g, expert))):
//    rank(i) = #{j : e_j==e_i and (g_j > g_i or (g_j==g_i and j<i))}.
//    Deterministic, no atomics; all-pairs scan staged through LDS.
// =====================================================================
#define RCH 1024
__global__ __launch_bounds__(256)
void rank_dispatch(const int* __restrict__ ids, const float* __restrict__ gates,
                   int* __restrict__ assign_pos, int* __restrict__ disp_tok,
                   int M, int cap)
{
  __shared__ int   sid[RCH];
  __shared__ float sg[RCH];
  int tid = threadIdx.x;
  int i   = blockIdx.x * blockDim.x + tid;
  int   e = (i < M) ? ids[i]   : -1;
  float g = (i < M) ? gates[i] : 0.f;

  int r = 0;
  for (int base = 0; base < M; base += RCH) {
    for (int j = tid; j < RCH; j += blockDim.x) {
      int jj = base + j;
      sid[j] = (jj < M) ? ids[jj]   : -1;
      sg[j]  = (jj < M) ? gates[jj] : 0.f;
    }
    __syncthreads();
    #pragma unroll 8
    for (int j = 0; j < RCH; ++j) {
      int jj = base + j;
      if (sid[j] == e && (sg[j] > g || (sg[j] == g && jj < i))) ++r;
    }
    __syncthreads();
  }
  if (i < M) {
    int pos = (r < cap) ? (e * cap + r) : (NEXP * cap);  // NEXP*cap = dropped
    assign_pos[i] = pos;
    if (r < cap) disp_tok[e * cap + r] = i >> 1;         // token id = i / TOPK
  }
}

// =====================================================================
// 4) One-time x -> bf16 cast so GEMM1's gathered A rows can be DMA'd
//    straight into LDS with async-to-LDS copies (no VGPR round trip).
// =====================================================================
__global__ void cast_x(const float* __restrict__ x, unsigned short* __restrict__ xb,
                       long long n4)
{
  long long i = blockIdx.x * (long long)blockDim.x + threadIdx.x;
  if (i < n4) {
    float4 v = ((const float4*)x)[i];
    uint2 pk;
    pk.x = pack2bf(v.x, v.y);
    pk.y = pack2bf(v.z, v.w);
    ((uint2*)xb)[i] = pk;
  }
}

// =====================================================================
// 5) Unified expert GEMM: out = act(A @ W + bias)
//    - A (bf16, global) copied into LDS with GLOBAL_LOAD_ASYNC_TO_LDS_B128
//      (ASYNCcnt-tracked), software-pipelined double buffer.
//    - W (f32, global) staged through VGPRs; pairs of k-rows are packed
//      with one v_cvt_pk_bf16_f32 per dword into transposed Bs[n][k].
//    - 256 threads = 8 waves (4x2), each wave 32x64 of a 128x128 tile,
//      8x v_wmma_f32_16x16x32_bf16 per K-step.
//    GATHER_A: row index via disp_tok (GEMM1); else dense rows (GEMM2).
//    SILU: bias+SiLU -> bf16 out; else bias -> f32 out.
// =====================================================================
template<int KDIM, bool GATHER_A, bool SILU>
__global__ __launch_bounds__(256)
void expert_gemm(const unsigned short* __restrict__ Abf,
                 const float* __restrict__ W,
                 const float* __restrict__ bias,
                 const int* __restrict__ disp_tok,
                 unsigned short* __restrict__ out_bf,
                 float* __restrict__ out_f32,
                 int cap, int nout)
{
  __shared__ __align__(16) unsigned short As[2][BM * LDT];
  __shared__ __align__(16) unsigned short Bs[2][BN * LDT];

  const int tid  = threadIdx.x;
  const int lane = tid & 31;
  const int wid  = tid >> 5;
  const int wm   = (wid & 3) * 32;
  const int wn   = (wid >> 2) * 64;
  const int half = lane >> 4;
  const int l15  = lane & 15;

  const int e    = blockIdx.z;
  const int row0 = blockIdx.y * BM;
  const int col0 = blockIdx.x * BN;

  const float* We = W    + (size_t)e * KDIM * nout;
  const float* be = bias + (size_t)e * nout;

  // ---- per-thread async-A descriptors (loop-invariant row pointers) ----
  // Each thread DMAs 2 x 16B per K-step: rows tid/4 and tid/4+64, 8 bf16 each.
  const unsigned short* aptr[2];
  unsigned lds_a[2];
  {
    const int cA = (tid & 3) * 8;
    #pragma unroll
    for (int j = 0; j < 2; ++j) {
      int r = (tid >> 2) + j * 64;
      int grow = row0 + r; if (grow >= cap) grow = cap - 1;   // clamp: rows >= cap are masked at store
      size_t arow = GATHER_A ? (size_t)disp_tok[e * cap + grow]
                             : (size_t)e * cap + grow;
      aptr[j]  = Abf + arow * KDIM + cA;
      lds_a[j] = (unsigned)(uintptr_t)&As[0][r * LDT + cA];   // low 32 bits = LDS byte offset
    }
  }
  const unsigned aBufStride = (unsigned)(BM * LDT * sizeof(unsigned short));

  // B staging registers: two adjacent k-rows (kb, kb+1) x 4 columns
  float4 breg[2][2];

  auto issueA = [&](int kk, int p) {
    #pragma unroll
    for (int j = 0; j < 2; ++j) {
      unsigned long long g = (unsigned long long)(uintptr_t)(aptr[j] + kk);
      unsigned l = lds_a[j] + (unsigned)p * aBufStride;
      asm volatile("global_load_async_to_lds_b128 %0, %1, off"
                   :: "v"(l), "v"(g) : "memory");
    }
  };
  auto loadB = [&](int kk) {
    #pragma unroll
    for (int j = 0; j < 2; ++j) {
      int kb = (tid >> 5) * 2 + j * 16;
      int c4 = (tid & 31) * 4;
      const float* p0 = &We[(size_t)(kk + kb) * nout + col0 + c4];
      breg[j][0] = *(const float4*)p0;
      breg[j][1] = *(const float4*)(p0 + nout);
    }
  };
  auto storeB = [&](int p) {
    unsigned short* bs = &Bs[p][0];
    #pragma unroll
    for (int j = 0; j < 2; ++j) {
      int kb = (tid >> 5) * 2 + j * 16;
      int c4 = (tid & 31) * 4;
      *(unsigned*)&bs[(c4 + 0) * LDT + kb] = pack2bf(breg[j][0].x, breg[j][1].x);
      *(unsigned*)&bs[(c4 + 1) * LDT + kb] = pack2bf(breg[j][0].y, breg[j][1].y);
      *(unsigned*)&bs[(c4 + 2) * LDT + kb] = pack2bf(breg[j][0].z, breg[j][1].z);
      *(unsigned*)&bs[(c4 + 3) * LDT + kb] = pack2bf(breg[j][0].w, breg[j][1].w);
    }
  };

  f32x8 acc[2][4];
  #pragma unroll
  for (int mt = 0; mt < 2; ++mt)
    #pragma unroll
    for (int nt = 0; nt < 4; ++nt)
      #pragma unroll
      for (int v = 0; v < 8; ++v) acc[mt][nt][v] = 0.f;

  constexpr int NIT = KDIM / BK;

  // ---- prologue: fill buffer 0 ----
  issueA(0, 0);
  loadB(0);
  storeB(0);

  for (int it = 0; it < NIT; ++it) {
    const int p = it & 1;
    // drain my async copies for tile `it`; barrier makes everyone's LDS
    // writes (async A + ds-stored B) visible before compute.
    asm volatile("s_wait_asynccnt 0x0" ::: "memory");
    __syncthreads();

    if (it + 1 < NIT) {
      issueA((it + 1) * BK, p ^ 1);   // DMA next A tile while we compute
      loadB((it + 1) * BK);           // next B tile into registers
      if (it + 2 < NIT)
        __builtin_prefetch(&We[(size_t)((it + 2) * BK + (tid >> 5) * 2) * nout + col0 + (tid & 31) * 4], 0, 0);
    }

    AFrag a[2], b[4];
    #pragma unroll
    for (int mt = 0; mt < 2; ++mt) {
      const unsigned short* ap = &As[p][(wm + mt * 16 + l15) * LDT];
      a[mt].q[0] = *(const uint4*)(ap + half * 8);        // K 0..7  / 8..15
      a[mt].q[1] = *(const uint4*)(ap + 16 + half * 8);   // K 16..23 / 24..31
    }
    #pragma unroll
    for (int nt = 0; nt < 4; ++nt) {
      const unsigned short* bp = &Bs[p][(wn + nt * 16 + l15) * LDT];
      b[nt].q[0] = *(const uint4*)(bp + half * 8);
      b[nt].q[1] = *(const uint4*)(bp + 16 + half * 8);
    }
    #pragma unroll
    for (int mt = 0; mt < 2; ++mt)
      #pragma unroll
      for (int nt = 0; nt < 4; ++nt)
        acc[mt][nt] = __builtin_amdgcn_wmma_f32_16x16x32_bf16(
            false, a[mt].v, false, b[nt].v, (short)0, acc[mt][nt], false, false);

    if (it + 1 < NIT) storeB(p ^ 1);  // convert+transpose next B into the other buffer
  }

  // ---- epilogue ----
  #pragma unroll
  for (int mt = 0; mt < 2; ++mt)
    #pragma unroll
    for (int nt = 0; nt < 4; ++nt) {
      int gn = col0 + wn + nt * 16 + l15;
      float bv = be[gn];
      #pragma unroll
      for (int v = 0; v < 8; ++v) {
        int gm = row0 + wm + mt * 16 + half * 8 + v;
        if (gm < cap) {
          float val = acc[mt][nt][v] + bv;
          if (SILU) {
            val = val / (1.f + __expf(-val));
            out_bf[((size_t)e * cap + gm) * nout + gn] = f2bf(val);
          } else {
            out_f32[((size_t)e * cap + gm) * nout + gn] = val;
          }
        }
      }
    }
}

// =====================================================================
// 6) Combine: y[n] = sum_k gate_k * out_buf[pos_k]  (deterministic gather;
//    dropped assignments have pos == E*cap and are skipped)
// =====================================================================
__global__ __launch_bounds__(256)
void combine(const float* __restrict__ out_buf, const int* __restrict__ assign_pos,
             const float* __restrict__ gates, float* __restrict__ y, int cap)
{
  int n  = blockIdx.x;
  int p0 = assign_pos[n * 2 + 0];
  int p1 = assign_pos[n * 2 + 1];
  float g0 = gates[n * 2 + 0];
  float g1 = gates[n * 2 + 1];
  int lim = NEXP * cap;
  for (int d = threadIdx.x; d < D_MODEL; d += blockDim.x) {
    float acc = 0.f;
    if (p0 < lim) acc += g0 * out_buf[(size_t)p0 * D_MODEL + d];
    if (p1 < lim) acc += g1 * out_buf[(size_t)p1 * D_MODEL + d];
    y[(size_t)n * D_MODEL + d] = acc;
  }
}

// =====================================================================
extern "C" void kernel_launch(void* const* d_in, const int* in_sizes, int n_in,
                              void* d_out, int out_size, void* d_ws, size_t ws_size,
                              hipStream_t stream)
{
  const float* x  = (const float*)d_in[0];
  const float* Wr = (const float*)d_in[1];
  const float* br = (const float*)d_in[2];
  const float* W1 = (const float*)d_in[3];
  const float* b1 = (const float*)d_in[4];
  const float* W2 = (const float*)d_in[5];
  const float* b2 = (const float*)d_in[6];
  float* y = (float*)d_out;

  const int N     = in_sizes[0] / D_MODEL;
  const int M     = N * TOPK;
  const int cap   = (12 * N + 10 * NEXP - 1) / (10 * NEXP);  // ceil(1.2*N/E)
  const int slots = NEXP * cap;

  // workspace carve-up (~276 MB total)
  char* ws = (char*)d_ws;
  size_t o = 0;
  int*   ids        = (int*)(ws + o);   o += (size_t)M * 4;        o = (o + 255) & ~(size_t)255;
  float* gates      = (float*)(ws + o); o += (size_t)M * 4;        o = (o + 255) & ~(size_t)255;
  int*   assign_pos = (int*)(ws + o);   o += (size_t)M * 4;        o = (o + 255) & ~(size_t)255;
  int*   disp_tok   = (int*)(ws + o);   o += (size_t)slots * 4;    o = (o + 255) & ~(size_t)255;
  unsigned short* xb = (unsigned short*)(ws + o);
  o += (size_t)N * D_MODEL * 2;                                   o = (o + 255) & ~(size_t)255;
  unsigned short* hbuf = (unsigned short*)(ws + o);
  o += (size_t)slots * H_FFN * 2;                                 o = (o + 255) & ~(size_t)255;
  float* out_buf    = (float*)(ws + o); o += (size_t)slots * D_MODEL * 4;
  (void)ws_size; (void)n_in; (void)out_size;

  router_topk  <<<dim3((N + 7) / 8),       dim3(256), 0, stream>>>(x, Wr, br, ids, gates, N);
  zero_disp    <<<dim3((slots + 255)/256), dim3(256), 0, stream>>>(disp_tok, slots);
  rank_dispatch<<<dim3((M + 255) / 256),   dim3(256), 0, stream>>>(ids, gates, assign_pos, disp_tok, M, cap);

  long long n4 = (long long)N * D_MODEL / 4;
  cast_x<<<dim3((unsigned)((n4 + 255) / 256)), dim3(256), 0, stream>>>(x, xb, n4);

  dim3 g1(H_FFN / BN, (cap + BM - 1) / BM, NEXP);
  expert_gemm<D_MODEL, true, true><<<g1, dim3(256), 0, stream>>>(
      xb, W1, b1, disp_tok, hbuf, nullptr, cap, H_FFN);

  dim3 g2(D_MODEL / BN, (cap + BM - 1) / BM, NEXP);
  expert_gemm<H_FFN, false, false><<<g2, dim3(256), 0, stream>>>(
      hbuf, W2, b2, disp_tok, nullptr, out_buf, cap, D_MODEL);

  combine<<<dim3(N), dim3(256), 0, stream>>>(out_buf, assign_pos, gates, y, cap);
}
